// ENDEC_54975581389154
// MI455X (gfx1250) — matrix operand order repaired and checked by reading, not compiled, not run on hardware
//
#include <hip/hip_runtime.h>

typedef unsigned short u16;
typedef __attribute__((ext_vector_type(16))) __bf16 bf16x16;
typedef __attribute__((ext_vector_type(8)))  float   f32x8;

#define BB 64
#define TEE 256
#define TDD 256
#define EE 256
#define HH 512
#define VV 128
#define NWG 32   // column-tile workgroups for RNN kernel

// ---------------- workspace layout (bytes) ----------------
#define OFF_WT_E0 0                         // 2048x768 bf16
#define OFF_WT_E1 (OFF_WT_E0 + 2048*768*2)  // 2048x1024
#define OFF_WT_D0 (OFF_WT_E1 + 2048*1024*2) // 2048x1280
#define OFF_WT_D1 (OFF_WT_D0 + 2048*1280*2) // 2048x1024
#define OFF_WT_P  (OFF_WT_D1 + 2048*1024*2) // 128x512
#define OFF_EMB   (OFF_WT_P  + 128*512*2)   // 128x256
#define OFF_H     (OFF_EMB   + 128*256*2)   // 2 layer * 2 parity * 64*512 bf16
#define OFF_ENL   (OFF_H     + 4*BB*HH*2)   // 64*512 bf16
#define OFF_DEC   (OFF_ENL   + BB*HH*2)     // 64*256*512 bf16
#define OFF_BAR   (OFF_DEC   + (size_t)BB*TDD*HH*2)

__device__ __forceinline__ u16 f2bf(float f) {
  unsigned u = __float_as_uint(f);
  unsigned r = u + 0x7FFFu + ((u >> 16) & 1u);
  return (u16)(r >> 16);
}

__device__ __forceinline__ bf16x16 frag16(const u16* p0, const u16* p1) {
  union { uint4 q[2]; bf16x16 v; } u;
  u.q[0] = *(const uint4*)p0;
  u.q[1] = *(const uint4*)p1;
  return u.v;
}

// fast path: v_exp_f32 + v_rcp_f32 (no IEEE-div Newton chains on the critical path)
__device__ __forceinline__ float sigm(float x) {
  return __builtin_amdgcn_rcpf(1.0f + __expf(-x));
}
__device__ __forceinline__ float tanh_(float x) { return 2.0f * sigm(2.0f * x) - 1.0f; }

// ---------------- grid barrier (32 co-resident WGs) ----------------
__device__ __forceinline__ void gbar(unsigned* bar) {
  __syncthreads();
  if (threadIdx.x == 0) {
    __threadfence();
    volatile unsigned* cnt = bar;
    volatile unsigned* gen = bar + 32;
    unsigned g = *gen;
    unsigned a = atomicAdd((unsigned*)cnt, 1u) + 1u;
    if (a == (unsigned)NWG) {
      *cnt = 0u;
      __threadfence();
      atomicAdd((unsigned*)gen, 1u);
    } else {
      while (*gen == g) { __builtin_amdgcn_s_sleep(1); }
    }
    __threadfence();
  }
  __syncthreads();
}

// ---------------- one contiguous K-segment of the gate GEMM ----------------
// A: 16x32 bf16/k-tile, lane row=l16, contiguous chunk [kbase..kbase+16)
// B: 32x16 bf16/k-tile, lane col=l16, contiguous chunk [khalf..khalf+16)
// b0..b3 are pre-offset: wt + (q*512+colC)*K + kstart + khalf
__device__ __forceinline__ void gemm_seg(
    int ktiles, const u16* __restrict__ pa_base, int kbase,
    const u16* __restrict__ b0, const u16* __restrict__ b1,
    const u16* __restrict__ b2, const u16* __restrict__ b3,
    f32x8& ai, f32x8& aj, f32x8& af, f32x8& ao)
{
#pragma unroll 1
  for (int kt = 0; kt < ktiles; ++kt) {
    const u16* pa = pa_base + kt * 32 + kbase;
    bf16x16 A = frag16(pa, pa + 8);
    const u16* p0 = b0 + kt * 32;
    const u16* p1 = b1 + kt * 32;
    const u16* p2 = b2 + kt * 32;
    const u16* p3 = b3 + kt * 32;
    ai = __builtin_amdgcn_wmma_f32_16x16x32_bf16(false, A, false, frag16(p0, p0 + 8), (short)0, ai, false, false);
    aj = __builtin_amdgcn_wmma_f32_16x16x32_bf16(false, A, false, frag16(p1, p1 + 8), (short)0, aj, false, false);
    af = __builtin_amdgcn_wmma_f32_16x16x32_bf16(false, A, false, frag16(p2, p2 + 8), (short)0, af, false, false);
    ao = __builtin_amdgcn_wmma_f32_16x16x32_bf16(false, A, false, frag16(p3, p3 + 8), (short)0, ao, false, false);
  }
}

__device__ __forceinline__ void gates(
    const f32x8& ai, const f32x8& aj, const f32x8& af, const f32x8& ao,
    int t, const int* len, float* c, float* h, float* nh_fresh)
{
#pragma unroll
  for (int r = 0; r < 8; ++r) {
    float nc = c[r] * sigm(af[r] + 1.0f) + sigm(ai[r]) * tanh_(aj[r]);
    float nh = tanh_(nc) * sigm(ao[r]);
    nh_fresh[r] = nh;
    if (t < len[r]) { c[r] = nc; h[r] = nh; }
  }
}

__device__ __forceinline__ void acc_init(f32x8& a, float b) {
#pragma unroll
  for (int r = 0; r < 8; ++r) a[r] = b;
}

// ---------------- persistent encoder+decoder recurrence ----------------
__global__ __launch_bounds__(128, 1)
void rnn_kernel(const int* __restrict__ en_input, const int* __restrict__ en_lengths,
                const int* __restrict__ dec_input, const int* __restrict__ dec_lengths,
                const float* __restrict__ encb0, const float* __restrict__ encb1,
                const float* __restrict__ decb0, const float* __restrict__ decb1,
                const u16* __restrict__ wt_e0, const u16* __restrict__ wt_e1,
                const u16* __restrict__ wt_d0, const u16* __restrict__ wt_d1,
                const u16* __restrict__ embed_bf,
                u16* __restrict__ hbufs, u16* __restrict__ en_last,
                u16* __restrict__ decout, unsigned* __restrict__ bar)
{
  const int g     = blockIdx.x;            // 0..31: owns col-tile g of every gate
  const int wave  = threadIdx.x >> 5;      // 0..3 : M-tile (16 batch rows)
  const int lane  = threadIdx.x & 31;
  const int l16   = lane & 15;
  const int lhalf = lane >> 4;
  const int colC  = g * 16 + l16;          // hidden-unit column 0..511
  const int rowA  = wave * 16 + l16;       // A-matrix row (batch)
  const int kbase = lhalf * 16;
  const int khalf = lhalf * 16;
  const int rowC0 = wave * 16 + lhalf * 8; // first C/D row of this lane

  // h state lives in ONE buffer: hbufs[(layer*2 + parity)*B*H + b*H + col].
  // Parity folded into integer offsets (no pointer selects -> global, not flat).
  const size_t aoff = (size_t)rowA * HH;   // A-row offset within a parity plane
  const size_t HP   = (size_t)BB * HH;     // parity plane stride

  // weight column pointers per gate, pre-offset by khalf (segment starts added at call sites)
  const u16* we0[4]; const u16* we1[4]; const u16* wd0[4]; const u16* wd1[4];
#pragma unroll
  for (int q = 0; q < 4; ++q) {
    we0[q] = wt_e0 + (size_t)(q * 512 + colC) * 768  + khalf;
    we1[q] = wt_e1 + (size_t)(q * 512 + colC) * 1024 + khalf;
    wd0[q] = wt_d0 + (size_t)(q * 512 + colC) * 1280 + khalf;
    wd1[q] = wt_d1 + (size_t)(q * 512 + colC) * 1024 + khalf;
  }
  float be0[4], be1[4], bd0[4], bd1[4];
#pragma unroll
  for (int q = 0; q < 4; ++q) {
    be0[q] = encb0[q * 512 + colC]; be1[q] = encb1[q * 512 + colC];
    bd0[q] = decb0[q * 512 + colC]; bd1[q] = decb1[q * 512 + colC];
  }
  int enl[8], dcl[8];
#pragma unroll
  for (int r = 0; r < 8; ++r) { enl[r] = en_lengths[rowC0 + r]; dcl[r] = dec_lengths[rowC0 + r]; }

  // persistent cell/hidden state in registers (carries enc_state into decoder)
  float c0[8], h0r[8], c1[8], h1r[8], nh[8];
#pragma unroll
  for (int r = 0; r < 8; ++r) { c0[r] = 0.f; h0r[r] = 0.f; c1[r] = 0.f; h1r[r] = 0.f; }

  f32x8 ai, aj, af, ao;

  // ---------------- encoder ----------------
  for (int t = 0; t < TEE; ++t) {
    const size_t rp = (size_t)(t & 1), wp = rp ^ 1;
    const int tok = en_input[rowA * TEE + t];
    // layer 0: A = [embed(tok) | h0(t-1)], K = 256 + 512
    acc_init(ai, be0[0]); acc_init(aj, be0[1]); acc_init(af, be0[2]); acc_init(ao, be0[3]);
    gemm_seg(8,  embed_bf + (size_t)tok * EE, kbase, we0[0],       we0[1],       we0[2],       we0[3],       ai, aj, af, ao);
    gemm_seg(16, hbufs + rp * HP + aoff,      kbase, we0[0] + 256, we0[1] + 256, we0[2] + 256, we0[3] + 256, ai, aj, af, ao);
    gates(ai, aj, af, ao, t, enl, c0, h0r, nh);
#pragma unroll
    for (int r = 0; r < 8; ++r) hbufs[wp * HP + (size_t)(rowC0 + r) * HH + colC] = f2bf(h0r[r]);
    gbar(bar);
    // layer 1: A = [h0(t) | h1(t-1)], K = 512 + 512
    acc_init(ai, be1[0]); acc_init(aj, be1[1]); acc_init(af, be1[2]); acc_init(ao, be1[3]);
    gemm_seg(16, hbufs + wp * HP + aoff,       kbase, we1[0],       we1[1],       we1[2],       we1[3],       ai, aj, af, ao);
    gemm_seg(16, hbufs + (2 + rp) * HP + aoff, kbase, we1[0] + 512, we1[1] + 512, we1[2] + 512, we1[3] + 512, ai, aj, af, ao);
    gates(ai, aj, af, ao, t, enl, c1, h1r, nh);
#pragma unroll
    for (int r = 0; r < 8; ++r) {
      hbufs[(2 + wp) * HP + (size_t)(rowC0 + r) * HH + colC] = f2bf(h1r[r]);
      if (t == enl[r] - 1) en_last[(size_t)(rowC0 + r) * HH + colC] = f2bf(nh[r]);
    }
    gbar(bar);
  }

  // ---------------- decoder (init state = enc final; parity continues) ----------------
  for (int t = 0; t < TDD; ++t) {
    const size_t rp = (size_t)(t & 1), wp = rp ^ 1;
    const int tok = dec_input[rowA * TDD + t];
    // layer 0: A = [embed(tok) | en_last | h0(t-1)], K = 256 + 512 + 512
    acc_init(ai, bd0[0]); acc_init(aj, bd0[1]); acc_init(af, bd0[2]); acc_init(ao, bd0[3]);
    gemm_seg(8,  embed_bf + (size_t)tok * EE, kbase, wd0[0],       wd0[1],       wd0[2],       wd0[3],       ai, aj, af, ao);
    gemm_seg(16, en_last + aoff,              kbase, wd0[0] + 256, wd0[1] + 256, wd0[2] + 256, wd0[3] + 256, ai, aj, af, ao);
    gemm_seg(16, hbufs + rp * HP + aoff,      kbase, wd0[0] + 768, wd0[1] + 768, wd0[2] + 768, wd0[3] + 768, ai, aj, af, ao);
    gates(ai, aj, af, ao, t, dcl, c0, h0r, nh);
#pragma unroll
    for (int r = 0; r < 8; ++r) hbufs[wp * HP + (size_t)(rowC0 + r) * HH + colC] = f2bf(h0r[r]);
    gbar(bar);
    // layer 1
    acc_init(ai, bd1[0]); acc_init(aj, bd1[1]); acc_init(af, bd1[2]); acc_init(ao, bd1[3]);
    gemm_seg(16, hbufs + wp * HP + aoff,       kbase, wd1[0],       wd1[1],       wd1[2],       wd1[3],       ai, aj, af, ao);
    gemm_seg(16, hbufs + (2 + rp) * HP + aoff, kbase, wd1[0] + 512, wd1[1] + 512, wd1[2] + 512, wd1[3] + 512, ai, aj, af, ao);
    gates(ai, aj, af, ao, t, dcl, c1, h1r, nh);
#pragma unroll
    for (int r = 0; r < 8; ++r) {
      hbufs[(2 + wp) * HP + (size_t)(rowC0 + r) * HH + colC] = f2bf(h1r[r]);
      decout[((size_t)(rowC0 + r) * TDD + t) * HH + colC] = (t < dcl[r]) ? f2bf(nh[r]) : (u16)0;
    }
    gbar(bar);
  }
}

// ---------------- projection: [B*TD x 512] @ [512 x 128] + projb ----------------
__global__ __launch_bounds__(128, 1)
void proj_kernel(const u16* __restrict__ decout, const u16* __restrict__ wt_p,
                 const float* __restrict__ projb, float* __restrict__ out)
{
  const int wave  = threadIdx.x >> 5;
  const int lane  = threadIdx.x & 31;
  const int l16   = lane & 15;
  const int lhalf = lane >> 4;
  const int mtile = blockIdx.x * 4 + wave;       // 0..1023
  const int rowA  = mtile * 16 + l16;            // row = b*TD + t
  const int rowC0 = mtile * 16 + lhalf * 8;
  const int kbase = lhalf * 16, khalf = lhalf * 16;

  f32x8 acc[8];
#pragma unroll
  for (int nt = 0; nt < 8; ++nt) {
    float bv = projb[nt * 16 + l16];
#pragma unroll
    for (int r = 0; r < 8; ++r) acc[nt][r] = bv;
  }
  const u16* arow = decout + (size_t)rowA * HH;
#pragma unroll 1
  for (int kt = 0; kt < 16; ++kt) {
    const u16* pa = arow + kt * 32 + kbase;
    bf16x16 A = frag16(pa, pa + 8);
#pragma unroll
    for (int nt = 0; nt < 8; ++nt) {
      const u16* pb = wt_p + (size_t)(nt * 16 + l16) * HH + kt * 32 + khalf;
      acc[nt] = __builtin_amdgcn_wmma_f32_16x16x32_bf16(false, A, false, frag16(pb, pb + 8), (short)0, acc[nt], false, false);
    }
  }
#pragma unroll
  for (int nt = 0; nt < 8; ++nt)
#pragma unroll
    for (int r = 0; r < 8; ++r)
      out[(size_t)(rowC0 + r) * VV + nt * 16 + l16] = acc[nt][r];
}

// ---------------- prep kernels ----------------
__global__ __launch_bounds__(256)
void k_wt_transpose(const float* __restrict__ src, u16* __restrict__ dst, int K, int N) {
  long total = (long)K * N;
  for (long i = (long)blockIdx.x * 256 + threadIdx.x; i < total; i += (long)gridDim.x * 256) {
    int n = (int)(i / K), k = (int)(i % K);
    dst[i] = f2bf(src[(size_t)k * N + n]);   // dst[n][k] = src[k][n]
  }
}
__global__ __launch_bounds__(256)
void k_conv(const float* __restrict__ src, u16* __restrict__ dst, int n) {
  for (int i = blockIdx.x * 256 + threadIdx.x; i < n; i += gridDim.x * 256) dst[i] = f2bf(src[i]);
}
__global__ __launch_bounds__(256)
void k_zero16(u16* __restrict__ dst, int n) {
  for (int i = blockIdx.x * 256 + threadIdx.x; i < n; i += gridDim.x * 256) dst[i] = 0;
}

extern "C" void kernel_launch(void* const* d_in, const int* in_sizes, int n_in,
                              void* d_out, int out_size, void* d_ws, size_t ws_size,
                              hipStream_t stream) {
  (void)in_sizes; (void)n_in; (void)out_size; (void)ws_size;
  const int*   en_input    = (const int*)d_in[0];
  const int*   en_lengths  = (const int*)d_in[1];
  const int*   dec_input   = (const int*)d_in[2];
  const int*   dec_lengths = (const int*)d_in[3];
  const float* embed       = (const float*)d_in[4];
  const float* encW0 = (const float*)d_in[5];  const float* encb0 = (const float*)d_in[6];
  const float* encW1 = (const float*)d_in[7];  const float* encb1 = (const float*)d_in[8];
  const float* decW0 = (const float*)d_in[9];  const float* decb0 = (const float*)d_in[10];
  const float* decW1 = (const float*)d_in[11]; const float* decb1 = (const float*)d_in[12];
  const float* projW = (const float*)d_in[13]; const float* projb = (const float*)d_in[14];
  float* out = (float*)d_out;

  char* ws = (char*)d_ws;
  u16* wt_e0    = (u16*)(ws + OFF_WT_E0);
  u16* wt_e1    = (u16*)(ws + OFF_WT_E1);
  u16* wt_d0    = (u16*)(ws + OFF_WT_D0);
  u16* wt_d1    = (u16*)(ws + OFF_WT_D1);
  u16* wt_p     = (u16*)(ws + OFF_WT_P);
  u16* embed_bf = (u16*)(ws + OFF_EMB);
  u16* hbufs    = (u16*)(ws + OFF_H);
  u16* en_last  = (u16*)(ws + OFF_ENL);
  u16* decout   = (u16*)(ws + OFF_DEC);
  unsigned* bar = (unsigned*)(ws + OFF_BAR);

  // prep: bf16-transpose weights, convert embed, zero state + barrier
  k_wt_transpose<<<1024, 256, 0, stream>>>(encW0, wt_e0, 768, 2048);
  k_wt_transpose<<<1024, 256, 0, stream>>>(encW1, wt_e1, 1024, 2048);
  k_wt_transpose<<<1024, 256, 0, stream>>>(decW0, wt_d0, 1280, 2048);
  k_wt_transpose<<<1024, 256, 0, stream>>>(decW1, wt_d1, 1024, 2048);
  k_wt_transpose<<<256, 256, 0, stream>>>(projW, wt_p, 512, 128);
  k_conv<<<128, 256, 0, stream>>>(embed, embed_bf, VV * EE);
  k_zero16<<<256, 256, 0, stream>>>(hbufs, 4 * BB * HH);
  k_zero16<<<1, 256, 0, stream>>>(en_last, BB * HH);
  k_zero16<<<1, 256, 0, stream>>>((u16*)bar, 128);

  // persistent recurrence (encoder + decoder), then projection
  rnn_kernel<<<NWG, 128, 0, stream>>>(en_input, en_lengths, dec_input, dec_lengths,
                                      encb0, encb1, decb0, decb1,
                                      wt_e0, wt_e1, wt_d0, wt_d1, embed_bf,
                                      hbufs, en_last, decout, bar);
  proj_kernel<<<256, 128, 0, stream>>>(decout, wt_p, projb, out);
}
